// long_window_LSTM_recurrentDropout_46677704573606
// MI455X (gfx1250) — matrix operand order repaired and checked by reading, not compiled, not run on hardware
//
#include <hip/hip_runtime.h>
#include <hip/hip_bf16.h>
#include <math.h>

// ---------------------------------------------------------------------------
// Problem constants
// ---------------------------------------------------------------------------
constexpr int Bn = 64;      // batch
constexpr int Tn = 512;     // timesteps
constexpr int Dn = 64;      // input dim
constexpr int Hn = 512;     // hidden dim
constexpr int Gn = 4 * Hn;  // 2048 gate columns
constexpr int On = 64;      // output dim
constexpr int NWG = 16;     // persistent workgroups (each owns 32 h-columns)
constexpr int NTHR = 512;   // threads per persistent WG (16 wave32s)
constexpr int LDSP = 132;   // LDS gate pitch (128 cols + pad)

typedef __bf16 bf16_t;
typedef __attribute__((ext_vector_type(16))) __bf16 v16bf;
typedef __attribute__((ext_vector_type(8)))  __bf16 v8bf;
typedef __attribute__((ext_vector_type(8)))  float  v8f;

// ---------------------------------------------------------------------------
// Workspace layout (bytes)
// ---------------------------------------------------------------------------
constexpr size_t OFF_WIH0P = 0;                               // 128*2*32*16 bf16
constexpr size_t OFF_WHH0P = OFF_WIH0P + 262144;              // 128*16*32*16 bf16
constexpr size_t OFF_WIH1P = OFF_WHH0P + 2097152;
constexpr size_t OFF_WHH1P = OFF_WIH1P + 2097152;
constexpr size_t OFF_B0SUM = OFF_WHH1P + 2097152;             // 2048 f32
constexpr size_t OFF_B1SUM = OFF_B0SUM + 8192;
constexpr size_t OFF_XBF   = OFF_B1SUM + 8192;                // 64*512*64 bf16
constexpr size_t OFF_H0    = OFF_XBF + 4194304;               // 64*512 bf16
constexpr size_t OFF_H1    = OFF_H0 + 65536;
constexpr size_t OFF_H1F32 = OFF_H1 + 65536;                  // 64*512 f32
constexpr size_t OFF_Y0    = OFF_H1F32 + 131072;              // 512*64*512 bf16
constexpr size_t OFF_BAR   = OFF_Y0 + 33554432;               // 2 u32

// ---------------------------------------------------------------------------
// Fragment loaders (layouts per CDNA5 ISA 7.12.2, wave32)
// A 16x32 bf16: lane L: m = L%16, two 8-elem K runs at k0+(L/16)*8 and +16.
// B 32x16 bf16 (pre-packed): lane L: n = L%16, same two 8-elem K runs as A.
// ---------------------------------------------------------------------------
__device__ __forceinline__ v16bf load_afrag(const bf16_t* base, int ld, int m0,
                                            int k0, int lane) {
    int m   = m0 + (lane & 15);
    int grp = lane >> 4;
    const bf16_t* p = base + (size_t)m * ld + k0 + grp * 8;
    union { v16bf v; v8bf h[2]; } u;
    u.h[0] = *(const v8bf*)(p);
    u.h[1] = *(const v8bf*)(p + 16);
    return u.v;
}

__device__ __forceinline__ v16bf load_bfrag(const bf16_t* packed, int ntg, int KT,
                                            int kt, int lane) {
    const bf16_t* p = packed + ((size_t)(ntg * KT + kt) * 32 + lane) * 16;
    union { v16bf v; v8bf h[2]; } u;
    u.h[0] = *(const v8bf*)(p);
    u.h[1] = *(const v8bf*)(p + 8);
    return u.v;
}

// One K-sweep of a GEMM part. Ping-pong software pipeline: two independent
// buffer sets, each written only by loads (no register rotation copies).
// KT is even (2 or 16). Each wave owns 2 M-tiles of N-tile ntg.
__device__ __forceinline__ void gemm_part(v8f acc[2], const bf16_t* act, int ld,
                                          int KT, const bf16_t* wp, int ntg,
                                          int m0base, int lane) {
    v16bf b0  = load_bfrag(wp, ntg, KT, 0, lane);
    v16bf a00 = load_afrag(act, ld, m0base, 0, lane);
    v16bf a01 = load_afrag(act, ld, m0base + 16, 0, lane);
    v16bf b1, a10, a11;
#pragma unroll 1
    for (int kt = 0; kt + 2 < KT; kt += 2) {
        // loads for kt+1 (buffer B) in flight over buffer-A WMMAs
        b1  = load_bfrag(wp, ntg, KT, kt + 1, lane);
        a10 = load_afrag(act, ld, m0base, (kt + 1) * 32, lane);
        a11 = load_afrag(act, ld, m0base + 16, (kt + 1) * 32, lane);
        acc[0] = __builtin_amdgcn_wmma_f32_16x16x32_bf16(
            false, a00, false, b0, (short)0, acc[0], false, false);
        acc[1] = __builtin_amdgcn_wmma_f32_16x16x32_bf16(
            false, a01, false, b0, (short)0, acc[1], false, false);
        // loads for kt+2 (buffer A) in flight over buffer-B WMMAs
        b0  = load_bfrag(wp, ntg, KT, kt + 2, lane);
        a00 = load_afrag(act, ld, m0base, (kt + 2) * 32, lane);
        a01 = load_afrag(act, ld, m0base + 16, (kt + 2) * 32, lane);
        acc[0] = __builtin_amdgcn_wmma_f32_16x16x32_bf16(
            false, a10, false, b1, (short)0, acc[0], false, false);
        acc[1] = __builtin_amdgcn_wmma_f32_16x16x32_bf16(
            false, a11, false, b1, (short)0, acc[1], false, false);
    }
    // epilogue: kt = KT-2 (already in buffer A) and kt = KT-1
    b1  = load_bfrag(wp, ntg, KT, KT - 1, lane);
    a10 = load_afrag(act, ld, m0base, (KT - 1) * 32, lane);
    a11 = load_afrag(act, ld, m0base + 16, (KT - 1) * 32, lane);
    acc[0] = __builtin_amdgcn_wmma_f32_16x16x32_bf16(
        false, a00, false, b0, (short)0, acc[0], false, false);
    acc[1] = __builtin_amdgcn_wmma_f32_16x16x32_bf16(
        false, a01, false, b0, (short)0, acc[1], false, false);
    acc[0] = __builtin_amdgcn_wmma_f32_16x16x32_bf16(
        false, a10, false, b1, (short)0, acc[0], false, false);
    acc[1] = __builtin_amdgcn_wmma_f32_16x16x32_bf16(
        false, a11, false, b1, (short)0, acc[1], false, false);
}

// C layout (16x16 f32): VGPR r -> M = r (lanes 0-15) / 8+r (lanes 16-31), N = lane%16
__device__ __forceinline__ void store_acc(float* gl, v8f acc[2], int nt, int mh,
                                          int lane) {
    int n     = nt * 16 + (lane & 15);
    int mbase = (lane >> 4) * 8;
#pragma unroll
    for (int mt = 0; mt < 2; ++mt)
#pragma unroll
        for (int r = 0; r < 8; ++r)
            gl[((mh * 2 + mt) * 16 + mbase + r) * LDSP + n] = acc[mt][r];
}

__device__ __forceinline__ float sigmoidf(float x) {
    return 1.0f / (1.0f + __expf(-x));
}

// Pointwise LSTM cell update; c lives in registers (4 rows per thread, 512 thr).
__device__ __forceinline__ void cell_update(const float* gl, const float* bsum,
                                            int wgid, int tid, float c[4],
                                            bf16_t* hbuf, float* hf32,
                                            bf16_t* ybuf) {
    int jj  = tid & 31;          // local h column within WG slice
    int bg  = tid >> 5;          // row group 0..15
    int col = 32 * wgid + jj;    // global h column
    float bi = bsum[col], bf = bsum[512 + col], bgg = bsum[1024 + col],
          bo = bsum[1536 + col];
#pragma unroll
    for (int r = 0; r < 4; ++r) {
        int b = bg * 4 + r;
        float gi = gl[b * LDSP + jj] + bi;
        float gf = gl[b * LDSP + 32 + jj] + bf;
        float gg = gl[b * LDSP + 64 + jj] + bgg;
        float go = gl[b * LDSP + 96 + jj] + bo;
        float i_ = sigmoidf(gi);
        float f_ = sigmoidf(gf);
        float g_ = tanhf(gg);
        float o_ = sigmoidf(go);
        c[r] = f_ * c[r] + i_ * g_;
        float h = o_ * tanhf(c[r]);
        hbuf[(size_t)b * Hn + col] = (bf16_t)h;
        if (ybuf) ybuf[(size_t)b * Hn + col] = (bf16_t)h;
        if (hf32) hf32[(size_t)b * Hn + col] = h;
    }
}

// Device-wide sense-reversing barrier (NWG co-resident WGs).
__device__ __forceinline__ void grid_barrier(unsigned* bar) {
    __threadfence();  // publish this WG's global stores at agent scope
    __syncthreads();
    if (threadIdx.x == 0) {
        unsigned* cnt = bar;
        unsigned* gen = bar + 1;
        unsigned g = __hip_atomic_load(gen, __ATOMIC_RELAXED, __HIP_MEMORY_SCOPE_AGENT);
        unsigned prev = __hip_atomic_fetch_add(cnt, 1u, __ATOMIC_ACQ_REL,
                                               __HIP_MEMORY_SCOPE_AGENT);
        if (prev == (unsigned)(NWG - 1)) {
            __hip_atomic_store(cnt, 0u, __ATOMIC_RELAXED, __HIP_MEMORY_SCOPE_AGENT);
            __hip_atomic_store(gen, g + 1u, __ATOMIC_RELEASE, __HIP_MEMORY_SCOPE_AGENT);
        } else {
            while (__hip_atomic_load(gen, __ATOMIC_ACQUIRE,
                                     __HIP_MEMORY_SCOPE_AGENT) == g) {
                __builtin_amdgcn_s_sleep(2);
            }
        }
    }
    __syncthreads();
}

// ---------------------------------------------------------------------------
// Prep kernels
// ---------------------------------------------------------------------------
__global__ void zero_init(unsigned* h0, unsigned* h1, unsigned* bar) {
    int i = blockIdx.x * blockDim.x + threadIdx.x;
    int stride = gridDim.x * blockDim.x;
    int n = (Bn * Hn) / 2;  // u32 words per bf16 h buffer
    for (int j = i; j < n; j += stride) { h0[j] = 0u; h1[j] = 0u; }
    if (i == 0) { bar[0] = 0u; bar[1] = 0u; }
}

__global__ void pack_x(const float* __restrict__ x, bf16_t* __restrict__ xbf, int n) {
    int i = blockIdx.x * blockDim.x + threadIdx.x;
    int stride = gridDim.x * blockDim.x;
    for (int j = i; j < n; j += stride) xbf[j] = (bf16_t)x[j];
}

// Pack W[N=2048, K] (row-major f32) into B-operand fragment order:
// dst[((ntg*KT + kt)*32 + lane)*16 + i] = W[ntg*16 + lane%16, kt*32 + (lane/16)*8 + (i/8)*16 + i%8]
__global__ void pack_w(const float* __restrict__ W, bf16_t* __restrict__ dst, int K) {
    int KT = K / 32;
    int total = 128 * KT * 32;
    int i = blockIdx.x * blockDim.x + threadIdx.x;
    int stride = gridDim.x * blockDim.x;
    for (int idx = i; idx < total; idx += stride) {
        int lane = idx & 31;
        int kt   = (idx >> 5) % KT;
        int ntg  = idx / (32 * KT);
        int n    = ntg * 16 + (lane & 15);
        int kb   = kt * 32 + (lane >> 4) * 8;
        const float* s = W + (size_t)n * K + kb;
        bf16_t* d = dst + (size_t)idx * 16;
#pragma unroll
        for (int e = 0; e < 16; ++e) d[e] = (bf16_t)s[(e & 7) + (e >> 3) * 16];
    }
}

__global__ void bias_sum(const float* __restrict__ bi, const float* __restrict__ bh,
                         float* __restrict__ out) {
    int i = blockIdx.x * blockDim.x + threadIdx.x;
    if (i < Gn) out[i] = bi[i] + bh[i];
}

// ---------------------------------------------------------------------------
// Persistent 2-layer LSTM kernel: NWG workgroups x 512 threads (16 wave32s)
// Wave w: N-tile nt = w/2, M-half mh = w%2 (M-tiles 2*mh, 2*mh+1).
// ---------------------------------------------------------------------------
__global__ void __launch_bounds__(NTHR, 1)
lstm_persistent(const bf16_t* __restrict__ xbf,
                const bf16_t* __restrict__ wih0p, const bf16_t* __restrict__ whh0p,
                const bf16_t* __restrict__ wih1p, const bf16_t* __restrict__ whh1p,
                const float* __restrict__ b0sum, const float* __restrict__ b1sum,
                bf16_t* h0, bf16_t* h1, float* h1f32, bf16_t* y0,
                unsigned* bar) {
    __shared__ float gl[64 * LDSP];  // gate exchange: 64 rows x 128 cols (+pad)

    const int wgid = blockIdx.x;
    const int tid  = threadIdx.x;
    const int lane = tid & 31;
    const int wv   = tid >> 5;        // 0..15
    const int nt   = wv >> 1;         // local N-tile 0..7
    const int mh   = wv & 1;          // M-half: tiles {2mh, 2mh+1}
    const int m0b  = mh * 32;         // first M row of this wave's tiles
    // Global packed-weight N-tile: gate = nt/2, 16-col tile within WG's 32-col slice
    const int ntg  = (nt >> 1) * 32 + 2 * wgid + (nt & 1);

    float c[4];
#pragma unroll
    for (int r = 0; r < 4; ++r) c[r] = 0.0f;

    // -------- layer 0 sweep --------
#pragma unroll 1
    for (int t = 0; t < Tn; ++t) {
        v8f acc[2] = {};
        gemm_part(acc, xbf + (size_t)t * Dn, Tn * Dn, Dn / 32, wih0p, ntg, m0b, lane);
        gemm_part(acc, h0, Hn, Hn / 32, whh0p, ntg, m0b, lane);
        store_acc(gl, acc, nt, mh, lane);
        __syncthreads();
        cell_update(gl, b0sum, wgid, tid, c, h0, nullptr,
                    y0 + (size_t)t * Bn * Hn);
        grid_barrier(bar);
    }

    // -------- carry: h1_init = h0_final (c carries in registers) --------
    {
        int jj = tid & 31, bg = tid >> 5, col = 32 * wgid + jj;
#pragma unroll
        for (int r = 0; r < 4; ++r) {
            int b = bg * 4 + r;
            h1[(size_t)b * Hn + col] = h0[(size_t)b * Hn + col];
        }
    }
    grid_barrier(bar);

    // -------- layer 1 sweep --------
#pragma unroll 1
    for (int t = 0; t < Tn; ++t) {
        v8f acc[2] = {};
        gemm_part(acc, y0 + (size_t)t * Bn * Hn, Hn, Hn / 32, wih1p, ntg, m0b, lane);
        gemm_part(acc, h1, Hn, Hn / 32, whh1p, ntg, m0b, lane);
        store_acc(gl, acc, nt, mh, lane);
        __syncthreads();
        cell_update(gl, b1sum, wgid, tid, c, h1, h1f32, nullptr);
        grid_barrier(bar);
    }
}

// ---------------------------------------------------------------------------
// Final projection: out[64,64] = h1 @ lin_w^T + lin_b  (fp32, tiny)
// ---------------------------------------------------------------------------
__global__ void final_proj(const float* __restrict__ h1f32,
                           const float* __restrict__ linw,
                           const float* __restrict__ linb,
                           float* __restrict__ out) {
    int g = blockIdx.x * blockDim.x + threadIdx.x;
    if (g >= Bn * On) return;
    int b = g / On, o = g % On;
    const float* hr = h1f32 + (size_t)b * Hn;
    const float* wr = linw + (size_t)o * Hn;
    float s = 0.0f;
    for (int k = 0; k < Hn; ++k) s = fmaf(hr[k], wr[k], s);
    out[g] = s + linb[o];
}

// ---------------------------------------------------------------------------
// Launch
// ---------------------------------------------------------------------------
extern "C" void kernel_launch(void* const* d_in, const int* in_sizes, int n_in,
                              void* d_out, int out_size, void* d_ws, size_t ws_size,
                              hipStream_t stream) {
    const float* x    = (const float*)d_in[0];
    const float* Wih0 = (const float*)d_in[1];
    const float* Whh0 = (const float*)d_in[2];
    const float* bih0 = (const float*)d_in[3];
    const float* bhh0 = (const float*)d_in[4];
    const float* Wih1 = (const float*)d_in[5];
    const float* Whh1 = (const float*)d_in[6];
    const float* bih1 = (const float*)d_in[7];
    const float* bhh1 = (const float*)d_in[8];
    const float* linw = (const float*)d_in[9];
    const float* linb = (const float*)d_in[10];

    char* ws = (char*)d_ws;
    bf16_t*   wih0p = (bf16_t*)(ws + OFF_WIH0P);
    bf16_t*   whh0p = (bf16_t*)(ws + OFF_WHH0P);
    bf16_t*   wih1p = (bf16_t*)(ws + OFF_WIH1P);
    bf16_t*   whh1p = (bf16_t*)(ws + OFF_WHH1P);
    float*    b0sum = (float*)(ws + OFF_B0SUM);
    float*    b1sum = (float*)(ws + OFF_B1SUM);
    bf16_t*   xbf   = (bf16_t*)(ws + OFF_XBF);
    bf16_t*   h0    = (bf16_t*)(ws + OFF_H0);
    bf16_t*   h1    = (bf16_t*)(ws + OFF_H1);
    float*    h1f32 = (float*)(ws + OFF_H1F32);
    bf16_t*   y0    = (bf16_t*)(ws + OFF_Y0);
    unsigned* bar   = (unsigned*)(ws + OFF_BAR);

    zero_init<<<64, 256, 0, stream>>>((unsigned*)h0, (unsigned*)h1, bar);
    pack_x<<<2048, 256, 0, stream>>>(x, xbf, Bn * Tn * Dn);
    pack_w<<<64, 256, 0, stream>>>(Wih0, wih0p, Dn);
    pack_w<<<256, 256, 0, stream>>>(Whh0, whh0p, Hn);
    pack_w<<<256, 256, 0, stream>>>(Wih1, wih1p, Hn);
    pack_w<<<256, 256, 0, stream>>>(Whh1, whh1p, Hn);
    bias_sum<<<8, 256, 0, stream>>>(bih0, bhh0, b0sum);
    bias_sum<<<8, 256, 0, stream>>>(bih1, bhh1, b1sum);

    lstm_persistent<<<NWG, NTHR, 0, stream>>>(xbf, wih0p, whh0p, wih1p, whh1p,
                                              b0sum, b1sum, h0, h1, h1f32, y0, bar);

    final_proj<<<16, 256, 0, stream>>>(h1f32, linw, linb, (float*)d_out);
}
